// CFTLayer2d_89541478187672
// MI455X (gfx1250) — matrix operand order repaired and checked by reading, not compiled, not run on hardware
//
#include <hip/hip_runtime.h>
#include <hip/hip_bf16.h>
#include <math.h>

// CDNA5 / gfx1250 implementation of the Chebyshev feature-transform layer.
// One workgroup (256 threads = 8 wave32) per (b, c, u, v) segment.
// The 6x6 coefficient matrix  coeffs[i][j] = sum_pixels (xn*T_i)*(T_j)
// is computed with v_wmma_f32_16x16x32_f16 (K = pixel dimension, 32 pixels
// per WMMA, accumulated over 16 chunks per wave, 8 waves reduced via LDS).

typedef _Float16 v16h __attribute__((ext_vector_type(16)));
typedef _Float16 v8h  __attribute__((ext_vector_type(8)));
typedef float    v8f  __attribute__((ext_vector_type(8)));

#define SEGS   4
#define HS     64
#define WSZ    64
#define CIN    64
#define COUT   64
#define NC     36          // m*m = 6*6 coefficient count
#define SEGPIX 4096        // 64*64 pixels per segment

// ---------------------------------------------------------------------------
// Pre-kernel: Wsum[(uv*36 + i)*64 + ci] = sum_co W[u,v,m1,m2,ci,co]
// (collapses the C_OUT reduction; layout of w is [S,S,6,6,CIN,COUT] so the
//  flattened (u,v,i,ci) index times 64 is exactly the row base in w).
// ---------------------------------------------------------------------------
__global__ void wsum_kernel(const float* __restrict__ w, float* __restrict__ wsum) {
    int idx = blockIdx.x * blockDim.x + threadIdx.x;
    if (idx >= SEGS * SEGS * NC * CIN) return;
    const float* p = w + (size_t)idx * COUT;
    float s = 0.0f;
#pragma unroll
    for (int o = 0; o < COUT; o += 4) {
        float4 v = *(const float4*)(p + o);
        s += v.x + v.y + v.z + v.w;
    }
    wsum[idx] = s;
}

// ---------------------------------------------------------------------------
// Main kernel: one block per segment.
// ---------------------------------------------------------------------------
__global__ __launch_bounds__(256) void cft_kernel(const float* __restrict__ x,
                                                  const float* __restrict__ wsum,
                                                  float* __restrict__ out) {
    __shared__ __align__(16) float     seg[SEGPIX];       // 16 KB segment cache
    __shared__ __align__(16) _Float16  mxMat[8][512];     // per-wave A staging: 16 rows x 32 K (f16)
    __shared__ __align__(16) _Float16  tMat[8][512];      // per-wave B staging: 16 rows x 32 K (f16)
    __shared__ float coeffW[8][NC];                       // per-wave coeff partials
    __shared__ float dotPart[NC];
    __shared__ float redMin[8], redMax[8];
    __shared__ float finals[4];                           // 0:min 1:max 3:signal

    const int t    = threadIdx.x;
    const int lane = t & 31;
    const int wid  = t >> 5;

    const int bid = blockIdx.x;
    const int uv  = bid & 15;            // u*4 + v
    const int v_  = uv & 3;
    const int u_  = uv >> 2;
    const int c   = (bid >> 4) & 63;
    const int b   = bid >> 10;

    const float* gbase = x   + ((((size_t)b * CIN + c) * 256 + (size_t)u_ * HS) * 256 + (size_t)v_ * WSZ);
    float*       obase = out + ((((size_t)b * CIN + c) * 256 + (size_t)u_ * HS) * 256 + (size_t)v_ * WSZ);

    // Zero staging matrices once (rows >= 6 stay zero => padded WMMA rows/cols).
    {
        int* z1 = (int*)&mxMat[0][0];
        int* z2 = (int*)&tMat[0][0];
        for (int i = t; i < 2048; i += 256) { z1[i] = 0; z2[i] = 0; }
    }

    // ---- Phase 1: coalesced load into LDS + min/max reduction ----
    float mn = 3.4e38f, mx = -3.4e38f;
#pragma unroll
    for (int k = 0; k < 4; ++k) {
        int p  = k * 1024 + t * 4;          // covers 0..4095, float4-aligned
        int hh = p >> 6, ww = p & 63;
        float4 v = *(const float4*)(gbase + hh * 256 + ww);
        *(float4*)(&seg[p]) = v;
        mn = fminf(mn, fminf(fminf(v.x, v.y), fminf(v.z, v.w)));
        mx = fmaxf(mx, fmaxf(fmaxf(v.x, v.y), fmaxf(v.z, v.w)));
    }
#pragma unroll
    for (int off = 16; off > 0; off >>= 1) {
        mn = fminf(mn, __shfl_xor(mn, off, 32));
        mx = fmaxf(mx, __shfl_xor(mx, off, 32));
    }
    if (lane == 0) { redMin[wid] = mn; redMax[wid] = mx; }
    __syncthreads();
    if (t == 0) {
        float a = redMin[0], bmax = redMax[0];
        for (int i = 1; i < 8; ++i) { a = fminf(a, redMin[i]); bmax = fmaxf(bmax, redMax[i]); }
        finals[0] = a; finals[1] = bmax;
    }
    __syncthreads();
    const float lo    = finals[0];
    const float scale = 2.0f / (finals[1] - lo + 1e-8f);

    // ---- Phase 2: WMMA accumulation of coeffs[i][j] over the pixel (K) axis ----
    _Float16* mrow = mxMat[wid];
    _Float16* trow = tMat[wid];

    // Operand addressing per CDNA5 ISA VGPR layouts (wave32):
    //  A (16x32 f16): lane holds row M=lane%16; halves = K 0..7,16..23 (lanes<16)
    //                 or K 8..15,24..31 (lanes>=16).
    //  B (32x16 f16): lane holds col N=lane%16; halves = K 0..7,16..23 (lanes<16)
    //                 or K 8..15,24..31 (lanes>=16), same as A.
    const int mh   = lane & 15;
    const int hsel = lane >> 4;
    const int aoff = mh * 32 + hsel * 8;
    const int boff = mh * 32 + hsel * 8;

    v8f acc = {0.f, 0.f, 0.f, 0.f, 0.f, 0.f, 0.f, 0.f};

    for (int k = 0; k < 16; ++k) {
        const int p  = (wid * 16 + k) * 32 + lane;   // this lane's pixel
        const float xv = seg[p];
        const float xn = (xv - lo) * scale - 1.0f;
        const float xc = fminf(fmaxf(xn, -1.0f + 1e-6f), 1.0f - 1e-6f);
        // Chebyshev recurrence: T_k(xc) == cos(k*acos(xc))
        const float x2 = xc + xc;
        const float t0 = 1.0f;
        const float t1 = xc;
        const float t2 = x2 * t1 - t0;
        const float t3 = x2 * t2 - t1;
        const float t4 = x2 * t3 - t2;
        const float t5 = x2 * t4 - t3;

        // Stage B rows (T_j) and A rows (xn * T_i), column = lane (= K index).
        trow[0 * 32 + lane] = (_Float16)t0;
        trow[1 * 32 + lane] = (_Float16)t1;
        trow[2 * 32 + lane] = (_Float16)t2;
        trow[3 * 32 + lane] = (_Float16)t3;
        trow[4 * 32 + lane] = (_Float16)t4;
        trow[5 * 32 + lane] = (_Float16)t5;
        mrow[0 * 32 + lane] = (_Float16)(xn * t0);
        mrow[1 * 32 + lane] = (_Float16)(xn * t1);
        mrow[2 * 32 + lane] = (_Float16)(xn * t2);
        mrow[3 * 32 + lane] = (_Float16)(xn * t3);
        mrow[4 * 32 + lane] = (_Float16)(xn * t4);
        mrow[5 * 32 + lane] = (_Float16)(xn * t5);

        // Same-wave LDS store->load ordering.
        asm volatile("s_wait_dscnt 0" ::: "memory");

        v8h a_lo = *(const v8h*)(mrow + aoff);
        v8h a_hi = *(const v8h*)(mrow + aoff + 16);
        v8h b_lo = *(const v8h*)(trow + boff);
        v8h b_hi = *(const v8h*)(trow + boff + 16);
        v16h A  = __builtin_shufflevector(a_lo, a_hi, 0,1,2,3,4,5,6,7,8,9,10,11,12,13,14,15);
        v16h Bm = __builtin_shufflevector(b_lo, b_hi, 0,1,2,3,4,5,6,7,8,9,10,11,12,13,14,15);

        acc = __builtin_amdgcn_wmma_f32_16x16x32_f16(
                  /*neg_a=*/false, A, /*neg_b=*/false, Bm,
                  /*c_mod=*/(short)0, acc, /*reuse_a=*/false, /*reuse_b=*/false);
    }

    // C tile layout: lane n (0..15) holds N=n, VGPR i (0..7) holds M=i.
    // Useful region: lanes 0..5 (j), elements 0..5 (i) -> coeff index i*6+j.
    if (lane < 6) {
        coeffW[wid][ 0 + lane] = acc[0];
        coeffW[wid][ 6 + lane] = acc[1];
        coeffW[wid][12 + lane] = acc[2];
        coeffW[wid][18 + lane] = acc[3];
        coeffW[wid][24 + lane] = acc[4];
        coeffW[wid][30 + lane] = acc[5];
    }
    __syncthreads();

    // ---- Phase 3: deterministic reduction, dot with Wsum, tanh ----
    if (t < NC) {
        float cs = 0.0f;
#pragma unroll
        for (int wv = 0; wv < 8; ++wv) cs += coeffW[wv][t];
        dotPart[t] = cs * wsum[(uv * NC + t) * CIN + c];
    }
    __syncthreads();
    if (t == 0) {
        float d = 0.0f;
        for (int i = 0; i < NC; ++i) d += dotPart[i];
        // coeffs were raw sums: /4096 (spatial mean) and /64 (C_OUT mean)
        finals[3] = tanhf(d * (1.0f / 262144.0f));
    }
    __syncthreads();

    const float s = finals[3];
    const float4 sv = make_float4(s, s, s, s);
#pragma unroll
    for (int k = 0; k < 4; ++k) {
        int p  = k * 1024 + t * 4;
        int hh = p >> 6, ww = p & 63;
        *(float4*)(obase + hh * 256 + ww) = sv;
    }
}

// ---------------------------------------------------------------------------
extern "C" void kernel_launch(void* const* d_in, const int* in_sizes, int n_in,
                              void* d_out, int out_size, void* d_ws, size_t ws_size,
                              hipStream_t stream) {
    const float* x = (const float*)d_in[0];
    const float* w = (const float*)d_in[1];
    float*       o = (float*)d_out;
    float*       wsum = (float*)d_ws;   // 4*4*36*64 floats = 144 KiB scratch

    // Precompute C_OUT-collapsed weights.
    wsum_kernel<<<(SEGS * SEGS * NC * CIN + 255) / 256, 256, 0, stream>>>(w, wsum);

    // One block per (b, c, u, v) segment: 4 * 64 * 16 = 4096 blocks.
    cft_kernel<<<4096, 256, 0, stream>>>(x, wsum, o);
}
